// SoftDTWLrp_395136991349
// MI455X (gfx1250) — matrix operand order repaired and checked by reading, not compiled, not run on hardware
//
#include <hip/hip_runtime.h>
#include <hip/hip_bf16.h>

// Problem constants from the reference
#define B_ 32
#define C_ 128
#define T_ 512
#define N_ 16
#define EPS_ 1e-6f
#define BIG_ 1e10f
// GAMMA == 1.0 so all /GAMMA factors are omitted.

typedef __attribute__((ext_vector_type(16))) __bf16 v16bf;
typedef __attribute__((ext_vector_type(8)))  float v8f;

__device__ inline float softmin2(float a, float b) {
    float m = fminf(a, b);
    return m - logf(expf(-(a - m)) + expf(-(b - m)));
}

// ---------------------------------------------------------------------------
// K0: zero scratch accumulators (Abar, Dbar, CiDir are atomically / additively
// updated, and the harness poisons ws with 0xAA).
// ---------------------------------------------------------------------------
__global__ void k0_zero(float* __restrict__ p, int count) {
    int idx = blockIdx.x * blockDim.x + threadIdx.x;
    if (idx < count) p[idx] = 0.0f;
}

// ---------------------------------------------------------------------------
// K1: delta[b,n,t] = |protos[n]|^2 + |Ai[b,:,t]|^2 - 2 * protos·Ai
// cross term via v_wmma_f32_16x16x32_bf16, K = C = 128 -> 4 WMMA per tile
// (fully unrolled).  One wave per (16-col t-tile, b).
// ---------------------------------------------------------------------------
__global__ void k1_delta(const float* __restrict__ Ai,
                         const float* __restrict__ protos,
                         float* __restrict__ delta) {
    const int t0 = blockIdx.x * 16;
    const int b  = blockIdx.y;
    const int lane = threadIdx.x;
    const int hi = lane >> 4;        // lane group (K-range selector)
    const int lo = lane & 15;
    const int t  = t0 + lo;

    __shared__ float p2s[16];
    if (lane < 16) {
        float s = 0.f;
        for (int k = 0; k < C_; ++k) { float p = protos[lane * C_ + k]; s += p * p; }
        p2s[lane] = s;
    }
    const float* arow = Ai + (size_t)b * C_ * T_;
    float x2 = 0.f;
    for (int c = 0; c < C_; ++c) { float v = arow[(size_t)c * T_ + t]; x2 += v * v; }
    __syncthreads();

    v8f acc = {};
    const int kgrpA = hi * 8;
    #pragma unroll
    for (int kc = 0; kc < 4; ++kc) {
        const int kbase = kc * 32;
        v16bf a, bb;
        #pragma unroll
        for (int j = 0; j < 16; ++j) {
            // A-matrix 16-bit layout: K = (j>>3)*16 + (lane>>4)*8 + (j&7), M = lane&15
            int KA = ((j >> 3) * 16) + kgrpA + (j & 7);
            a[j] = (__bf16)protos[lo * C_ + kbase + KA];
            // B-matrix 16-bit layout: K = (j>>3)*16 + (lane>>4)*8 + (j&7), N = lane&15
            int KB = ((j >> 3) * 16) + hi * 8 + (j & 7);
            bb[j] = (__bf16)arow[(size_t)(kbase + KB) * T_ + t];
        }
        acc = __builtin_amdgcn_wmma_f32_16x16x32_bf16(
            false, a, false, bb, (short)0, acc, false, false);
    }

    float* drow = delta + (size_t)b * N_ * T_;
    #pragma unroll
    for (int r = 0; r < 8; ++r) {
        int m = r + 8 * hi;                       // C/D layout: M = r + 8*(lane>>4)
        drow[(size_t)m * T_ + t] = p2s[m] + x2 - 2.0f * acc[r];
    }
}

// ---------------------------------------------------------------------------
// K2: soft-DTW forward DP (R) + backward pass (E).  One wave per batch b,
// lane i = DTW state (0..15); lanes >=16 shadow state 15 harmlessly.
// delta is staged into LDS (transposed, [T][16]) with coalesced loads so the
// serial dependence chain only touches LDS; R lives in LDS and is mirrored to
// global for K4; E is a scan carry, streamed straight to global.
// Dynamic LDS: 64 KB (Rl + Dl), well within 320 KB/WGP.
// ---------------------------------------------------------------------------
__global__ void k2_dp(const float* __restrict__ delta,
                      float* __restrict__ Rg,
                      float* __restrict__ Eg) {
    const int b = blockIdx.x;
    const int i = threadIdx.x;
    const int ic = (i < 15) ? i : 15;
    extern __shared__ float lds2[];
    float* Rl = lds2;             // [T][16]
    float* Dl = lds2 + T_ * 16;   // [T][16] transposed delta

    const float* del  = delta + (size_t)b * N_ * T_;
    float*       Rrow = Rg    + (size_t)b * N_ * T_;
    float*       Erow = Eg    + (size_t)b * N_ * T_;

    // stage delta[b] (row-major [16][T]) into LDS transposed [T][16]
    for (int idx = i; idx < N_ * T_; idx += 32) {
        int row = idx >> 9;            // / T_
        int col = idx & (T_ - 1);      // % T_
        Dl[col * 16 + row] = del[idx];
    }
    __syncthreads();

    // forward: R[0][i] = (i==0)? d[0][0] : BIG ; R[j] = d[j] + softmin(R[j-1], shift)
    float Rprev = (ic == 0) ? Dl[0] : BIG_;
    if (i < 16) { Rl[0 * 16 + i] = Rprev; Rrow[(size_t)i * T_ + 0] = Rprev; }
    for (int j = 1; j < T_; ++j) {
        float sh = __shfl_up(Rprev, 1);
        if (i == 0) sh = BIG_;
        float dj = Dl[j * 16 + ic];
        float Rj = dj + softmin2(Rprev, sh);
        if (i < 16) { Rl[j * 16 + i] = Rj; Rrow[(size_t)i * T_ + j] = Rj; }
        Rprev = Rj;
    }
    __syncthreads();

    // backward: E[T-1] = one-hot(n-1); E[j] = E[j+1]*w_stay + up(E[j+1])*w_adv
    float En = (ic == 15) ? 1.0f : 0.0f;
    if (i < 16) Erow[(size_t)i * T_ + (T_ - 1)] = En;
    for (int j = T_ - 2; j >= 0; --j) {
        float Rc = Rl[j * 16 + ic];
        float Rn = Rl[(j + 1) * 16 + ic];
        float dn = Dl[(j + 1) * 16 + ic];
        float Rn_up = __shfl_down(Rn, 1);
        float dn_up = __shfl_down(dn, 1);
        float En_up = __shfl_down(En, 1);
        if (ic == 15) { Rn_up = -BIG_; dn_up = 0.0f; En_up = 0.0f; }
        float Ec = En * expf(Rn - dn - Rc) + En_up * expf(Rn_up - dn_up - Rc);
        if (i < 16) Erow[(size_t)i * T_ + j] = Ec;
        En = Ec;
    }
}

// ---------------------------------------------------------------------------
// K3: out3 max + argmax, S = Rj/(out3+eps), scatter direct grad + Abar.
// block = (n, b), thread = c (128 threads).
// ---------------------------------------------------------------------------
__global__ void k3_pool(const float* __restrict__ Ai,
                        const float* __restrict__ Rj,
                        const float* __restrict__ Eg,
                        float* __restrict__ CiDir,
                        float* __restrict__ Abar) {
    const int ni = blockIdx.x;
    const int b  = blockIdx.y;
    const int c  = threadIdx.x;
    const float* Erow = Eg + (size_t)b * N_ * T_ + (size_t)ni * T_;
    const float* arow = Ai + (size_t)b * C_ * T_ + (size_t)c * T_;
    float best = -1.0f; int tb = 0;
    for (int t = 0; t < T_; ++t) {
        float v = arow[t] * Erow[t];
        if (v > best) { best = v; tb = t; }
    }
    float S = Rj[(size_t)b * C_ * N_ + (size_t)c * N_ + ni] / (best + EPS_);
    atomicAdd(&CiDir[(size_t)b * C_ * T_ + (size_t)c * T_ + tb], S * Erow[tb]);
    atomicAdd(&Abar [(size_t)b * N_ * T_ + (size_t)ni * T_ + tb], S * arow[tb]);
}

// ---------------------------------------------------------------------------
// K4: second-order soft-DTW — VJP of sdtw_align w.r.t. delta given Abar.
// Pass A = adjoint of the backward (E) scan, run forward in time.
// Pass B = adjoint of the forward (R) scan, run backward in time.
// delta, R, E are staged into LDS (transposed) so both serial chains read
// only LDS; Rbar also lives in LDS.  Dynamic LDS: 128 KB.
// ---------------------------------------------------------------------------
__global__ void k4_adjoint(const float* __restrict__ delta,
                           const float* __restrict__ Rg,
                           const float* __restrict__ Eg,
                           const float* __restrict__ Abar,
                           float* __restrict__ Dbar) {
    const int b = blockIdx.x;
    const int i = threadIdx.x;
    const int ic = (i < 15) ? i : 15;
    extern __shared__ float lds4[];
    float* Rbar = lds4;               // [T][16]
    float* Dl   = lds4 + 1 * T_ * 16; // [T][16] delta
    float* Rl   = lds4 + 2 * T_ * 16; // [T][16] R
    float* El   = lds4 + 3 * T_ * 16; // [T][16] E

    const float* del  = delta + (size_t)b * N_ * T_;
    const float* Rrow = Rg    + (size_t)b * N_ * T_;
    const float* Erow = Eg    + (size_t)b * N_ * T_;
    const float* Ab   = Abar  + (size_t)b * N_ * T_;
    float*       Db   = Dbar  + (size_t)b * N_ * T_;

    for (int idx = i; idx < N_ * T_; idx += 32) {
        int row = idx >> 9;
        int col = idx & (T_ - 1);
        Dl[col * 16 + row] = del[idx];
        Rl[col * 16 + row] = Rrow[idx];
        El[col * 16 + row] = Erow[idx];
        Rbar[idx] = 0.0f;
    }
    __syncthreads();

    // ---- Pass A: Ebar carried forward in time ----
    float cb = 0.0f;   // adjoint flowing into E[j] from step j-1
    for (int j = 0; j < T_ - 1; ++j) {
        float eb  = ((i < 16) ? Ab[(size_t)ic * T_ + j] : 0.0f) + cb;
        float Rc  = Rl[j * 16 + ic];
        float Rn  = Rl[(j + 1) * 16 + ic];
        float dn  = Dl[(j + 1) * 16 + ic];
        float Enx = El[(j + 1) * 16 + ic];
        float Rn_up = __shfl_down(Rn, 1);
        float dn_up = __shfl_down(dn, 1);
        float En_up = __shfl_down(Enx, 1);
        if (ic == 15) { Rn_up = -BIG_; dn_up = 0.0f; En_up = 0.0f; }
        float wst = expf(Rn - dn - Rc);
        float wad = expf(Rn_up - dn_up - Rc);
        float qs = eb * Enx  * wst;
        float qa = eb * En_up * wad;
        float ebw_up = __shfl_up(eb * wad, 1);
        float qa_up  = __shfl_up(qa, 1);
        if (i == 0) { ebw_up = 0.0f; qa_up = 0.0f; }
        cb = eb * wst + ebw_up;          // Ebar[j+1][i] += eb[i]*ws + eb[i-1]*wa[i-1]
        if (i < 16) {
            float r1 = qs + qa_up;       // contribution to Rbar[j+1] / -dbar[j+1]
            Rbar[(j + 1) * 16 + i] += r1;
            Db[(size_t)i * T_ + j + 1] = -r1;   // first (only) pass-A write to this slot
            Rbar[j * 16 + i] -= (qs + qa);
        }
    }
    __syncthreads();

    // ---- Pass B: adjoint of forward softmin recursion, backward in time ----
    for (int j = T_ - 1; j >= 1; --j) {
        float rb  = (i < 16) ? Rbar[j * 16 + i] : 0.0f;
        float Rjv = Rl[j * 16 + ic];
        float djv = Dl[j * 16 + ic];
        float Rp  = Rl[(j - 1) * 16 + ic];
        float Rpsh = __shfl_up(Rp, 1);
        if (i == 0) Rpsh = BIG_;
        float pa = expf(Rjv - djv - Rp);     // softmin weight (stay)
        float pb = expf(Rjv - djv - Rpsh);   // softmin weight (advance)
        float rbpb_dn = __shfl_down(rb * pb, 1);
        if (ic == 15) rbpb_dn = 0.0f;
        if (i < 16) {
            Db[(size_t)i * T_ + j] += rb;                     // dbar[j] += Rbar[j]
            Rbar[(j - 1) * 16 + i] += rb * pa + rbpb_dn;      // propagate to R[j-1]
        }
    }
    if (i == 0) Db[0] += Rbar[0];   // R0[0] = d[0][0]; other states were BIG consts
}

// ---------------------------------------------------------------------------
// K5: Ci = CiDir + 2*Ai*sum_n(dbar) - 2*protos^T·dbar ; out = Ai*Ci.
// protos^T·dbar is a [16 x n=16] @ [n=16 x 16] tile — one bf16 WMMA with
// K zero-padded from 16 to 32.
// ---------------------------------------------------------------------------
__global__ void k5_final(const float* __restrict__ Ai,
                         const float* __restrict__ protos,
                         const float* __restrict__ Dbar,
                         const float* __restrict__ CiDir,
                         float* __restrict__ out) {
    const int t0 = blockIdx.x * 16;
    const int c0 = blockIdx.y * 16;
    const int b  = blockIdx.z;
    const int lane = threadIdx.x;
    const int hi = lane >> 4;
    const int lo = lane & 15;
    const int t  = t0 + lo;
    const float* Db = Dbar + (size_t)b * N_ * T_;

    v16bf a, bb;
    #pragma unroll
    for (int j = 0; j < 16; ++j) {
        int KA = ((j >> 3) * 16) + hi * 8 + (j & 7);
        a[j]  = (KA < 16) ? (__bf16)protos[KA * C_ + (c0 + lo)] : (__bf16)0.0f;
        int KB = ((j >> 3) * 16) + hi * 8 + (j & 7);
        bb[j] = (KB < 16) ? (__bf16)Db[(size_t)KB * T_ + t]     : (__bf16)0.0f;
    }
    v8f acc = {};
    acc = __builtin_amdgcn_wmma_f32_16x16x32_bf16(
        false, a, false, bb, (short)0, acc, false, false);

    float sd = 0.0f;
    #pragma unroll
    for (int nn = 0; nn < N_; ++nn) sd += Db[(size_t)nn * T_ + t];

    const float* arow = Ai    + (size_t)b * C_ * T_;
    const float* crow = CiDir + (size_t)b * C_ * T_;
    float*       orow = out   + (size_t)b * C_ * T_;
    #pragma unroll
    for (int r = 0; r < 8; ++r) {
        int cc = c0 + r + 8 * hi;
        float ai = arow[(size_t)cc * T_ + t];
        float ci = crow[(size_t)cc * T_ + t] + 2.0f * ai * sd - 2.0f * acc[r];
        orow[(size_t)cc * T_ + t] = ai * ci;
    }
}

// ---------------------------------------------------------------------------
extern "C" void kernel_launch(void* const* d_in, const int* in_sizes, int n_in,
                              void* d_out, int out_size, void* d_ws, size_t ws_size,
                              hipStream_t stream) {
    const float* Rj     = (const float*)d_in[0];   // [B, C, n]
    const float* Ai     = (const float*)d_in[1];   // [B, C, 1, T] == [B, C, T]
    const float* protos = (const float*)d_in[2];   // [n, C]
    float* out = (float*)d_out;                    // [B, C, 1, T]
    float* ws  = (float*)d_ws;

    const size_t nT  = (size_t)B_ * N_ * T_;       // 262144 floats
    const size_t cT  = (size_t)B_ * C_ * T_;       // 2097152 floats
    float* delta = ws;                             // [B,n,T]
    float* Rg    = ws + 1 * nT;                    // [B,n,T]
    float* Eg    = ws + 2 * nT;                    // [B,n,T]  (= alignment matrix A)
    float* Abar  = ws + 3 * nT;                    // [B,n,T]
    float* Dbar  = ws + 4 * nT;                    // [B,n,T]  (= deltabar)
    float* CiDir = ws + 5 * nT;                    // [B,C,T]

    // zero the additively-updated buffers: Abar, Dbar, CiDir (contiguous)
    const int zcount = (int)(2 * nT + cT);
    k0_zero<<<(zcount + 255) / 256, 256, 0, stream>>>(Abar, zcount);

    const size_t lds2_bytes = 2 * (size_t)T_ * 16 * sizeof(float);  //  64 KB
    const size_t lds4_bytes = 4 * (size_t)T_ * 16 * sizeof(float);  // 128 KB

    k1_delta  <<<dim3(T_ / 16, B_),          32, 0,          stream>>>(Ai, protos, delta);
    k2_dp     <<<B_,                         32, lds2_bytes, stream>>>(delta, Rg, Eg);
    k3_pool   <<<dim3(N_, B_),               C_, 0,          stream>>>(Ai, Rj, Eg, CiDir, Abar);
    k4_adjoint<<<B_,                         32, lds4_bytes, stream>>>(delta, Rg, Eg, Abar, Dbar);
    k5_final  <<<dim3(T_ / 16, C_ / 16, B_), 32, 0,          stream>>>(Ai, protos, Dbar, CiDir, out);
}
